// GNN_80796924772944
// MI455X (gfx1250) — matrix operand order repaired and checked by reading, not compiled, not run on hardware
//
#include <hip/hip_runtime.h>

typedef _Float16 half_t;
typedef __attribute__((ext_vector_type(16))) _Float16 v16h;
typedef __attribute__((ext_vector_type(8)))  _Float16 v8h;
typedef __attribute__((ext_vector_type(8)))  float    v8f;

#define NODES 8192
#define KNN   32

__device__ __forceinline__ float lrelu(float v) { return v >= 0.f ? v : 0.2f * v; }

// A fragment: f16 16x32 (MxK). lane 0-15: M=lane, K = ko+{0..7,16..23};
// lane 16-31: M=lane-16, K = ko+{8..15,24..31}.  sA row-major, stride lda halfs.
__device__ __forceinline__ v16h load_a_frag(const half_t* sA, int lda, int mBase, int ko, int lane) {
    int m  = mBase + (lane & 15);
    int kb = (lane & 16) ? 8 : 0;
    const half_t* p = sA + m * lda + ko + kb;
    v8h lo = *(const v8h*)(p);        // K = ko+kb .. +7
    v8h hi = *(const v8h*)(p + 16);   // K = ko+kb+16 .. +23
    return __builtin_shufflevector(lo, hi, 0,1,2,3,4,5,6,7,8,9,10,11,12,13,14,15);
}

// B fragment: f16 32x16 (KxN). lane holds N=lane&15; K = ko+{0..7,16..23} (lane<16) / ko+{8..15,24..31}.
// sWt stored transposed [n][k] (stride ldk halfs) so the 16 K-values are contiguous.
__device__ __forceinline__ v16h load_b_frag(const half_t* sWt, int ldk, int nBase, int ko, int lane) {
    int n  = nBase + (lane & 15);
    int kg = (lane & 16) ? 8 : 0;
    const half_t* p = sWt + n * ldk + ko + kg;
    v8h lo = *(const v8h*)(p);
    v8h hi = *(const v8h*)(p + 16);
    return __builtin_shufflevector(lo, hi, 0,1,2,3,4,5,6,7,8,9,10,11,12,13,14,15);
}

#define WMMA_F16(A, B, C) \
    __builtin_amdgcn_wmma_f32_16x16x32_f16(false, (A), false, (B), (short)0, (C), false, false)

// ---------------------------------------------------------------------------
// Weight pre-pass: f32 row-major W[K][N] -> f16 transposed Wt[N][KP], zero-pad
// rows k >= K.  Makes per-block chunk staging a contiguous vector copy.
// ---------------------------------------------------------------------------
__global__ __launch_bounds__(256) void convert_wt_kernel(const float* __restrict__ W,
                                                         half_t* __restrict__ Wt,
                                                         int K, int KP, int N)
{
    int idx = blockIdx.x * 256 + threadIdx.x;
    if (idx >= N * KP) return;
    int n = idx / KP, k = idx - n * KP;
    Wt[idx] = (half_t)((k < K) ? W[k * N + n] : 0.f);
}

// ---------------------------------------------------------------------------
// EdgeConv: per block, 4 target nodes => 128 edges, 256 threads (8 waves).
// feat = cat[x_i, x_j - x_i] (2D) -> norm1+lrelu -> @W1 -> norm2+lrelu -> @W2
// out[i] = (1/32) * sum over i's 32 edges.
// ---------------------------------------------------------------------------
template <int D, int DH, int DO, bool BN>
__global__ __launch_bounds__(256) void edgeconv_kernel(
    const float* __restrict__ hin, const int* __restrict__ src,
    const float* __restrict__ g1, const float* __restrict__ b1, const half_t* __restrict__ Wt1,
    const float* __restrict__ g2, const float* __restrict__ b2, const half_t* __restrict__ Wt2,
    float* __restrict__ hout)
{
    constexpr int D2   = 2 * D;
    constexpr int KP   = (D2 + 31) & ~31;              // padded K (power of two here)
    constexpr int CMAX = (KP > DH) ? KP : DH;
    constexpr int LDA  = CMAX + 8;                     // skewed strides: kill bank conflicts
    constexpr int CMY  = (CMAX < 64) ? 64 : CMAX;
    constexpr int LDY  = CMY + 5;
    constexpr int LDW1 = KP + 8;
    constexpr int LDW2 = DH + 8;
    constexpr int LDWX = (LDW1 > LDW2) ? LDW1 : LDW2;

    __shared__ alignas(32) half_t sA [128 * LDA];      // A1 then A2 (f16)
    __shared__ alignas(32) half_t sWt[64 * LDWX];      // weight chunk, transposed [n][k]
    __shared__ alignas(16) float  sY [128 * LDY];      // feats -> Y -> Z (f32)
    __shared__ float sMu[128], sRs[128];
    __shared__ int   sSrc[128];

    const int tid = threadIdx.x, lane = tid & 31, wave = tid >> 5;
    const int n0 = blockIdx.x * 4;
    const float bnf = rsqrtf(1.0f + 1e-5f);

    if (tid < 128) sSrc[tid] = src[(n0 + (tid >> 5)) * KNN + (tid & 31)];
    __syncthreads();

    // ---- stage raw edge features (f32) into sY
    for (int idx = tid; idx < 128 * KP; idx += 256) {
        int r = idx / KP, c = idx % KP;
        int node = n0 + (r >> 5);
        float v = 0.f;
        if (c < D)        v = hin[node * D + c];
        else if (c < D2)  { int cc = c - D; v = hin[sSrc[r] * D + cc] - hin[node * D + cc]; }
        sY[r * LDY + c] = v;
    }
    __syncthreads();

    if (!BN) {
        if (tid < 128) {
            float s = 0.f, s2 = 0.f;
            for (int c = 0; c < D2; ++c) { float v = sY[tid * LDY + c]; s += v; s2 += v * v; }
            float mu = s * (1.f / D2);
            sMu[tid] = mu;
            sRs[tid] = rsqrtf(fmaxf(s2 * (1.f / D2) - mu * mu, 0.f) + 1e-5f);
        }
        __syncthreads();
    }
    for (int idx = tid; idx < 128 * KP; idx += 256) {
        int r = idx / KP, c = idx % KP;
        float v = 0.f;
        if (c < D2) {
            v = sY[r * LDY + c];
            v = BN ? (v * (g1[c] * bnf) + b1[c]) : ((v - sMu[r]) * sRs[r] * g1[c] + b1[c]);
            v = lrelu(v);
        }
        sA[r * LDA + c] = (half_t)v;
    }
    __syncthreads();

    // ---- GEMM1: Y[128xDH] = A1[128xKP] @ W1[KPxDH], 64-col chunks
    for (int nc = 0; nc < DH; nc += 64) {
        const v8h* gw = (const v8h*)(Wt1 + nc * KP);   // contiguous 64*KP halfs
        for (int i8 = tid; i8 < 64 * (KP / 8); i8 += 256) {
            int n = i8 / (KP / 8), k8 = i8 % (KP / 8);
            *(v8h*)(sWt + n * LDW1 + k8 * 8) = gw[i8];
        }
        __syncthreads();
        v8f a0 = {}, a1 = {}, a2 = {}, a3 = {};
        for (int ko = 0; ko < KP; ko += 32) {
            v16h af = load_a_frag(sA, LDA, wave * 16, ko, lane);
            a0 = WMMA_F16(af, load_b_frag(sWt, LDW1,  0, ko, lane), a0);
            a1 = WMMA_F16(af, load_b_frag(sWt, LDW1, 16, ko, lane), a1);
            a2 = WMMA_F16(af, load_b_frag(sWt, LDW1, 32, ko, lane), a2);
            a3 = WMMA_F16(af, load_b_frag(sWt, LDW1, 48, ko, lane), a3);
        }
        int mOff = wave * 16 + ((lane >> 4) << 3);
        int nOff = nc + (lane & 15);
        #pragma unroll
        for (int r = 0; r < 8; ++r) {
            sY[(mOff + r) * LDY + nOff +  0] = a0[r];
            sY[(mOff + r) * LDY + nOff + 16] = a1[r];
            sY[(mOff + r) * LDY + nOff + 32] = a2[r];
            sY[(mOff + r) * LDY + nOff + 48] = a3[r];
        }
        __syncthreads();
    }

    // ---- norm2 + lrelu on Y -> A2 (f16) in sA
    if (!BN) {
        if (tid < 128) {
            float s = 0.f, s2 = 0.f;
            for (int c = 0; c < DH; ++c) { float v = sY[tid * LDY + c]; s += v; s2 += v * v; }
            float mu = s * (1.f / DH);
            sMu[tid] = mu;
            sRs[tid] = rsqrtf(fmaxf(s2 * (1.f / DH) - mu * mu, 0.f) + 1e-5f);
        }
        __syncthreads();
    }
    for (int idx = tid; idx < 128 * DH; idx += 256) {
        int r = idx / DH, c = idx % DH;
        float v = sY[r * LDY + c];
        v = BN ? (v * (g2[c] * bnf) + b2[c]) : ((v - sMu[r]) * sRs[r] * g2[c] + b2[c]);
        sA[r * LDA + c] = (half_t)lrelu(v);
    }
    __syncthreads();

    // ---- GEMM2 + per-node aggregation, 64-col chunks of DO
    for (int nc = 0; nc < DO; nc += 64) {
        const v8h* gw = (const v8h*)(Wt2 + nc * DH);
        for (int i8 = tid; i8 < 64 * (DH / 8); i8 += 256) {
            int n = i8 / (DH / 8), k8 = i8 % (DH / 8);
            *(v8h*)(sWt + n * LDW2 + k8 * 8) = gw[i8];
        }
        __syncthreads();
        v8f a0 = {}, a1 = {}, a2 = {}, a3 = {};
        for (int ko = 0; ko < DH; ko += 32) {
            v16h af = load_a_frag(sA, LDA, wave * 16, ko, lane);
            a0 = WMMA_F16(af, load_b_frag(sWt, LDW2,  0, ko, lane), a0);
            a1 = WMMA_F16(af, load_b_frag(sWt, LDW2, 16, ko, lane), a1);
            a2 = WMMA_F16(af, load_b_frag(sWt, LDW2, 32, ko, lane), a2);
            a3 = WMMA_F16(af, load_b_frag(sWt, LDW2, 48, ko, lane), a3);
        }
        int mOff = wave * 16 + ((lane >> 4) << 3);
        int nOff = lane & 15;
        #pragma unroll
        for (int r = 0; r < 8; ++r) {
            sY[(mOff + r) * LDY + nOff +  0] = a0[r];
            sY[(mOff + r) * LDY + nOff + 16] = a1[r];
            sY[(mOff + r) * LDY + nOff + 32] = a2[r];
            sY[(mOff + r) * LDY + nOff + 48] = a3[r];
        }
        __syncthreads();
        int node = tid >> 6, col = tid & 63;          // 4 nodes x 64 cols = 256 threads
        float s = 0.f;
        #pragma unroll 4
        for (int r = 0; r < 32; ++r) s += sY[(node * 32 + r) * LDY + col];
        hout[(n0 + node) * DO + nc + col] = s * (1.f / 32.f);
        __syncthreads();
    }
}

// ---------------------------------------------------------------------------
// jk: per node, LN(concat[h1,h2,h3,h4]) -> lrelu -> @ W[512x1024]
// 128 nodes per block, 256 threads (8 waves).
// ---------------------------------------------------------------------------
__global__ __launch_bounds__(256) void jk_kernel(
    const float* __restrict__ h1, const float* __restrict__ h2,
    const float* __restrict__ h3, const float* __restrict__ h4,
    const float* __restrict__ g, const float* __restrict__ b,
    const half_t* __restrict__ Wt, float* __restrict__ out)
{
    constexpr int LDA = 512 + 8;
    constexpr int LDW = 512 + 8;
    __shared__ alignas(32) half_t sA [128 * LDA];
    __shared__ alignas(32) half_t sWt[64 * LDW];
    __shared__ float sMu[128], sRs[128];
    const int tid = threadIdx.x, lane = tid & 31, wave = tid >> 5;
    const int node0 = blockIdx.x * 128;

    if (tid < 128) {
        int n = node0 + tid;
        float s = 0.f, s2 = 0.f;
        for (int c = 0; c < 64;  ++c) { float v = h1[n * 64  + c]; s += v; s2 += v * v; }
        for (int c = 0; c < 64;  ++c) { float v = h2[n * 64  + c]; s += v; s2 += v * v; }
        for (int c = 0; c < 128; ++c) { float v = h3[n * 128 + c]; s += v; s2 += v * v; }
        for (int c = 0; c < 256; ++c) { float v = h4[n * 256 + c]; s += v; s2 += v * v; }
        float mu = s * (1.f / 512.f);
        sMu[tid] = mu;
        sRs[tid] = rsqrtf(fmaxf(s2 * (1.f / 512.f) - mu * mu, 0.f) + 1e-5f);
    }
    __syncthreads();
    for (int idx = tid; idx < 128 * 512; idx += 256) {
        int r = idx >> 9, c = idx & 511;
        int n = node0 + r;
        float v = (c < 64)  ? h1[n * 64 + c]
                : (c < 128) ? h2[n * 64 + (c - 64)]
                : (c < 256) ? h3[n * 128 + (c - 128)]
                :             h4[n * 256 + (c - 256)];
        v = (v - sMu[r]) * sRs[r] * g[c] + b[c];
        sA[r * LDA + c] = (half_t)lrelu(v);
    }
    __syncthreads();

    for (int nc = 0; nc < 1024; nc += 64) {
        const v8h* gw = (const v8h*)(Wt + nc * 512);
        for (int i8 = tid; i8 < 64 * 64; i8 += 256) {
            int n = i8 >> 6, k8 = i8 & 63;
            *(v8h*)(sWt + n * LDW + k8 * 8) = gw[i8];
        }
        __syncthreads();
        v8f a0 = {}, a1 = {}, a2 = {}, a3 = {};
        for (int ko = 0; ko < 512; ko += 32) {
            v16h af = load_a_frag(sA, LDA, wave * 16, ko, lane);
            a0 = WMMA_F16(af, load_b_frag(sWt, LDW,  0, ko, lane), a0);
            a1 = WMMA_F16(af, load_b_frag(sWt, LDW, 16, ko, lane), a1);
            a2 = WMMA_F16(af, load_b_frag(sWt, LDW, 32, ko, lane), a2);
            a3 = WMMA_F16(af, load_b_frag(sWt, LDW, 48, ko, lane), a3);
        }
        int mOff = wave * 16 + ((lane >> 4) << 3);
        int nOff = nc + (lane & 15);
        #pragma unroll
        for (int r = 0; r < 8; ++r) {
            out[(node0 + mOff + r) * 1024 + nOff +  0] = a0[r];
            out[(node0 + mOff + r) * 1024 + nOff + 16] = a1[r];
            out[(node0 + mOff + r) * 1024 + nOff + 32] = a2[r];
            out[(node0 + mOff + r) * 1024 + nOff + 48] = a3[r];
        }
        __syncthreads();
    }
}

// ---------------------------------------------------------------------------
// global max pool: [8192x1024] -> [4x1024]  (2048 nodes per graph)
// ---------------------------------------------------------------------------
__global__ __launch_bounds__(256) void pool_kernel(const float* __restrict__ jk,
                                                   float* __restrict__ pooled)
{
    int gph = blockIdx.x >> 2;
    int c = ((blockIdx.x & 3) << 8) + threadIdx.x;
    const float* p = jk + (size_t)(gph * 2048) * 1024 + c;
    float m = -3.402823466e38f;
    for (int n = 0; n < 2048; ++n) m = fmaxf(m, p[(size_t)n * 1024]);
    pooled[gph * 1024 + c] = m;
}

// ---------------------------------------------------------------------------
// head: three BN(eval)+lrelu+Linear+bias blocks on [4x1024]; tiny, one block.
// ---------------------------------------------------------------------------
__global__ __launch_bounds__(256) void head_kernel(
    const float* __restrict__ pooled,
    const float* __restrict__ g1, const float* __restrict__ b1,
    const float* __restrict__ W1, const float* __restrict__ bi1,
    const float* __restrict__ g2, const float* __restrict__ b2,
    const float* __restrict__ W2, const float* __restrict__ bi2,
    const float* __restrict__ g3, const float* __restrict__ b3,
    const float* __restrict__ W3, const float* __restrict__ bi3,
    float* __restrict__ out)
{
    __shared__ float act[4 * 1024];
    __shared__ float t1[4 * 512];
    __shared__ float t2[4 * 256];
    const int tid = threadIdx.x;
    const float bnf = rsqrtf(1.0f + 1e-5f);

    for (int idx = tid; idx < 4 * 1024; idx += 256) {
        int c = idx & 1023;
        act[idx] = lrelu(pooled[idx] * (g1[c] * bnf) + b1[c]);
    }
    __syncthreads();
    for (int idx = tid; idx < 4 * 512; idx += 256) {
        int bb = idx >> 9, n = idx & 511;
        float s = bi1[n];
        for (int k = 0; k < 1024; ++k) s += act[bb * 1024 + k] * W1[k * 512 + n];
        t1[idx] = s;
    }
    __syncthreads();
    for (int idx = tid; idx < 4 * 512; idx += 256) {
        int n = idx & 511;
        t1[idx] = lrelu(t1[idx] * (g2[n] * bnf) + b2[n]);
    }
    __syncthreads();
    for (int idx = tid; idx < 4 * 256; idx += 256) {
        int bb = idx >> 8, n = idx & 255;
        float s = bi2[n];
        for (int k = 0; k < 512; ++k) s += t1[bb * 512 + k] * W2[k * 256 + n];
        t2[idx] = s;
    }
    __syncthreads();
    for (int idx = tid; idx < 4 * 256; idx += 256) {
        int n = idx & 255;
        t2[idx] = lrelu(t2[idx] * (g3[n] * bnf) + b3[n]);
    }
    __syncthreads();
    for (int idx = tid; idx < 4 * 40; idx += 256) {
        int bb = idx / 40, n = idx % 40;
        float s = bi3[n];
        for (int k = 0; k < 256; ++k) s += t2[bb * 256 + k] * W3[k * 40 + n];
        out[idx] = s;
    }
}

// ---------------------------------------------------------------------------
extern "C" void kernel_launch(void* const* d_in, const int* in_sizes, int n_in,
                              void* d_out, int out_size, void* d_ws, size_t ws_size,
                              hipStream_t stream)
{
    (void)in_sizes; (void)n_in; (void)out_size; (void)ws_size;
    const float* x   = (const float*)d_in[0];
    const int*   src = (const int*)d_in[1];   // edge_index row 0 (sources); dst implied
    const float *c1g1=(const float*)d_in[3],  *c1b1=(const float*)d_in[4],  *c1W1=(const float*)d_in[5],
                *c1g2=(const float*)d_in[6],  *c1b2=(const float*)d_in[7],  *c1W2=(const float*)d_in[8];
    const float *c2g1=(const float*)d_in[9],  *c2b1=(const float*)d_in[10], *c2W1=(const float*)d_in[11],
                *c2g2=(const float*)d_in[12], *c2b2=(const float*)d_in[13], *c2W2=(const float*)d_in[14];
    const float *c3g1=(const float*)d_in[15], *c3b1=(const float*)d_in[16], *c3W1=(const float*)d_in[17],
                *c3g2=(const float*)d_in[18], *c3b2=(const float*)d_in[19], *c3W2=(const float*)d_in[20];
    const float *c4g1=(const float*)d_in[21], *c4b1=(const float*)d_in[22], *c4W1=(const float*)d_in[23],
                *c4g2=(const float*)d_in[24], *c4b2=(const float*)d_in[25], *c4W2=(const float*)d_in[26];
    const float *jkg=(const float*)d_in[27], *jkb=(const float*)d_in[28], *jkW=(const float*)d_in[29];
    const float *o1g=(const float*)d_in[30], *o1b=(const float*)d_in[31],
                *o1W=(const float*)d_in[32], *o1bi=(const float*)d_in[33];
    const float *o2g=(const float*)d_in[34], *o2b=(const float*)d_in[35],
                *o2W=(const float*)d_in[36], *o2bi=(const float*)d_in[37];
    const float *o3g=(const float*)d_in[38], *o3b=(const float*)d_in[39],
                *o3W=(const float*)d_in[40], *o3bi=(const float*)d_in[41];

    float* h1 = (float*)d_ws;              // 8192*64
    float* h2 = h1 + NODES * 64;           // 8192*64
    float* h3 = h2 + NODES * 64;           // 8192*128
    float* h4 = h3 + NODES * 128;          // 8192*256
    float* jko = h4 + NODES * 256;         // 8192*1024
    float* pooled = jko + NODES * 1024;    // 4*1024

    // f16 pre-transposed weights (Wt[N][KP]), 16B-aligned offsets
    half_t* w11 = (half_t*)(pooled + 4 * 1024);
    half_t* w12 = w11 + 64 * 32;
    half_t* w21 = w12 + 64 * 64;
    half_t* w22 = w21 + 64 * 128;
    half_t* w31 = w22 + 64 * 64;
    half_t* w32 = w31 + 128 * 128;
    half_t* w41 = w32 + 128 * 128;
    half_t* w42 = w41 + 256 * 256;
    half_t* wjk = w42 + 256 * 256;

    auto cvt = [&](const float* W, half_t* Wt, int K, int KP, int N) {
        int tot = N * KP;
        convert_wt_kernel<<<(tot + 255) / 256, 256, 0, stream>>>(W, Wt, K, KP, N);
    };
    cvt(c1W1, w11,  12,  32,   64);
    cvt(c1W2, w12,  64,  64,   64);
    cvt(c2W1, w21, 128, 128,   64);
    cvt(c2W2, w22,  64,  64,   64);
    cvt(c3W1, w31, 128, 128,  128);
    cvt(c3W2, w32, 128, 128,  128);
    cvt(c4W1, w41, 256, 256,  256);
    cvt(c4W2, w42, 256, 256,  256);
    cvt(jkW,  wjk, 512, 512, 1024);

    dim3 blk(256);
    edgeconv_kernel<6,   64,  64, true ><<<NODES / 4, blk, 0, stream>>>(x,  src, c1g1,c1b1,w11, c1g2,c1b2,w12, h1);
    edgeconv_kernel<64,  64,  64, false><<<NODES / 4, blk, 0, stream>>>(h1, src, c2g1,c2b1,w21, c2g2,c2b2,w22, h2);
    edgeconv_kernel<64,  128, 128,false><<<NODES / 4, blk, 0, stream>>>(h2, src, c3g1,c3b1,w31, c3g2,c3b2,w32, h3);
    edgeconv_kernel<128, 256, 256,false><<<NODES / 4, blk, 0, stream>>>(h3, src, c4g1,c4b1,w41, c4g2,c4b2,w42, h4);
    jk_kernel<<<NODES / 128, blk, 0, stream>>>(h1, h2, h3, h4, jkg, jkb, wjk, jko);
    pool_kernel<<<16, 256, 0, stream>>>(jko, pooled);
    head_kernel<<<1, 256, 0, stream>>>(pooled, o1g,o1b,o1W,o1bi, o2g,o2b,o2W,o2bi, o3g,o3b,o3W,o3bi,
                                       (float*)d_out);
}
